// SPDMatrix_2396591751300
// MI455X (gfx1250) — compile-verified, matrix operand order checked by reading, not run on hardware
//
#include <hip/hip_runtime.h>

// ---- problem constants (match reference) ----
#define BATCH 64
#define MROWS 512      // M_OUT
#define KDIM  1024     // N_FEAT
#define TILE  128      // output tile edge per workgroup
#define KC    32       // K per WMMA step (bf16)
#define LROW  40       // padded LDS row length in bf16 elems (64B data + 16B pad)

typedef __attribute__((ext_vector_type(16))) __bf16 v16bf;
typedef __attribute__((ext_vector_type(8)))  __bf16 v8bf;
typedef __attribute__((ext_vector_type(4)))  __bf16 v4bf;
typedef __attribute__((ext_vector_type(8)))  float  v8f;

typedef __attribute__((ext_vector_type(4))) unsigned int tdm_v4u;
typedef __attribute__((ext_vector_type(8))) int          tdm_v8i;
typedef __attribute__((ext_vector_type(4))) int          tdm_v4i;

// Build a 16x32 bf16 A/B fragment for one lane from LDS:
// lane group 0 (lanes 0-15):  K = [0..7]  and [16..23]
// lane group 1 (lanes 16-31): K = [8..15] and [24..31]
static __device__ __forceinline__ v16bf load_frag(const __bf16* p) {
  v8bf c0 = *(const v8bf*)(p);        // K base .. base+7
  v8bf c1 = *(const v8bf*)(p + 16);   // K base+16 .. base+23
  return __builtin_shufflevector(c0, c1, 0, 1, 2, 3, 4, 5, 6, 7,
                                         8, 9, 10, 11, 12, 13, 14, 15);
}

// ---------------------------------------------------------------------------
// TDM: one 2-D tile load, 128 rows x 32 bf16, row pitch KDIM elements,
// landing in LDS with 16B padding after every 64B row (=> LROW layout).
// D# per CDNA5 ISA.  6-arg builtin form (clang-23).
// ---------------------------------------------------------------------------
static __device__ __forceinline__ void tdm_load_tile(const __bf16* gsrc,
                                                     unsigned lds_off) {
  unsigned long long ga = (unsigned long long)(uintptr_t)gsrc;
  // group0: count=1 | lds_addr | global_addr[56:0] | type=2
  tdm_v4u g0 = { 1u, lds_off, (unsigned)ga,
                 (unsigned)((ga >> 32) & 0x1FFFFFFull) | (2u << 30) };
  // group1 fields
  const unsigned flags = (1u << 16)   // data_size = 2 bytes
                       | (1u << 20)   // pad_enable
                       | (3u << 22)   // pad_interval: 16 dwords (64B)
                       | (3u << 25);  // pad_amount: 4 dwords (16B)
  const unsigned TD0 = KC;            // tensor_dim0 (elements, OOB bound)
  const unsigned TD1 = TILE;          // tensor_dim1
  const unsigned long long S0 = KDIM; // tensor_dim0_stride (elements)
  tdm_v8i g1 = { (int)flags,
                 (int)((TD0 & 0xFFFFu) << 16),                        // td0 lo
                 (int)(((TD0 >> 16) & 0xFFFFu) | ((TD1 & 0xFFFFu) << 16)),
                 (int)(((TD1 >> 16) & 0xFFFFu) | ((unsigned)KC << 16)), // tile_dim0
                 (int)(TILE & 0xFFFFu),                               // tile_dim1
                 (int)(unsigned)(S0 & 0xFFFFFFFFull),
                 (int)(unsigned)((S0 >> 32) & 0xFFFFull),
                 0 };
  tdm_v4i g2 = { 0, 0, 0, 0 };
  tdm_v4i g3 = { 0, 0, 0, 0 };
  tdm_v8i g4 = { 0, 0, 0, 0, 0, 0, 0, 0 };
  __builtin_amdgcn_tensor_load_to_lds(g0, g1, g2, g3, g4, 0);
}

// ---------------------------------------------------------------------------
// Phase 1: fp32 -> (bf16 hi, bf16 lo) split + exact fp32 row norms.
// One wave per row; 4096 blocks x 256 threads.
// ---------------------------------------------------------------------------
__global__ __launch_bounds__(256)
void split_convert_kernel(const float* __restrict__ x,
                          __bf16* __restrict__ xhi,
                          __bf16* __restrict__ xlo,
                          float* __restrict__ sq) {
  const int row  = blockIdx.x * 8 + (threadIdx.x >> 5);
  const int lane = threadIdx.x & 31;
  const float* src = x + (size_t)row * KDIM + lane * 4;
  __bf16* dh = xhi + (size_t)row * KDIM + lane * 4;
  __bf16* dl = xlo + (size_t)row * KDIM + lane * 4;

  float s = 0.f;
#pragma unroll
  for (int k = 0; k < KDIM; k += 128) {   // 32 lanes x 4 floats per pass
    float4 f4 = *(const float4*)(src + k);
    float f[4] = {f4.x, f4.y, f4.z, f4.w};
    v4bf h, l;
#pragma unroll
    for (int e = 0; e < 4; ++e) {
      __bf16 hb = (__bf16)f[e];
      h[e] = hb;
      l[e] = (__bf16)(f[e] - (float)hb);
      s += f[e] * f[e];
    }
    *(v4bf*)(dh + k) = h;
    *(v4bf*)(dl + k) = l;
  }
#pragma unroll
  for (int off = 16; off > 0; off >>= 1) s += __shfl_down(s, off);
  if (lane == 0) sq[row] = s;
}

// ---------------------------------------------------------------------------
// Phase 2: batched split-bf16 WMMA GEMM, TDM-staged, double-buffered LDS.
// ---------------------------------------------------------------------------
__global__ __launch_bounds__(256, 2)
void spd_rbf_wmma_tdm_kernel(const __bf16* __restrict__ xhi,
                             const __bf16* __restrict__ xlo,
                             const float* __restrict__ sq,
                             float* __restrict__ out) {
  __shared__ __attribute__((aligned(16))) __bf16 sAhi[2][TILE * LROW];
  __shared__ __attribute__((aligned(16))) __bf16 sAlo[2][TILE * LROW];
  __shared__ __attribute__((aligned(16))) __bf16 sBhi[2][TILE * LROW];
  __shared__ __attribute__((aligned(16))) __bf16 sBlo[2][TILE * LROW];
  __shared__ float sSqA[TILE];
  __shared__ float sSqB[TILE];

  const int b   = blockIdx.y;
  const int tm  = blockIdx.x >> 2;
  const int tn  = blockIdx.x & 3;
  const int tid = threadIdx.x;
  const int i0  = tm * TILE;
  const int j0  = tn * TILE;

  // row norms for the epilogue (written before any barrier; loop barriers
  // guarantee visibility by the time the epilogue runs)
  if (tid < TILE)       sSqA[tid] = sq[b * MROWS + i0 + tid];
  else                  sSqB[tid - TILE] = sq[b * MROWS + j0 + (tid - TILE)];

  const __bf16* gAhi = xhi + ((size_t)(b * MROWS + i0)) * KDIM;
  const __bf16* gAlo = xlo + ((size_t)(b * MROWS + i0)) * KDIM;
  const __bf16* gBhi = xhi + ((size_t)(b * MROWS + j0)) * KDIM;
  const __bf16* gBlo = xlo + ((size_t)(b * MROWS + j0)) * KDIM;

  const int wave = tid >> 5;
  const int lane = tid & 31;
  const int lg   = lane >> 4;
  const int lr   = lane & 15;

  v8f acc[8];
#pragma unroll
  for (int t = 0; t < 8; ++t) acc[t] = {};

  // prologue: wave 0 launches the 4 TDM tile loads for chunk 0
  if (wave == 0) {
    tdm_load_tile(gAhi, (unsigned)(uintptr_t)&sAhi[0][0]);
    tdm_load_tile(gAlo, (unsigned)(uintptr_t)&sAlo[0][0]);
    tdm_load_tile(gBhi, (unsigned)(uintptr_t)&sBhi[0][0]);
    tdm_load_tile(gBlo, (unsigned)(uintptr_t)&sBlo[0][0]);
  }

  const int NSTEP = KDIM / KC;
  for (int n = 0; n < NSTEP; ++n) {
    const int buf = n & 1;
    if (wave == 0) __builtin_amdgcn_s_wait_tensorcnt(0);  // chunk n landed
    __syncthreads();  // all waves: chunk n visible; buf^1 free for reuse

    if (wave == 0 && (n + 1) < NSTEP) {   // prefetch chunk n+1 via TDM
      const size_t koff = (size_t)(n + 1) * KC;
      const int nb = buf ^ 1;
      tdm_load_tile(gAhi + koff, (unsigned)(uintptr_t)&sAhi[nb][0]);
      tdm_load_tile(gAlo + koff, (unsigned)(uintptr_t)&sAlo[nb][0]);
      tdm_load_tile(gBhi + koff, (unsigned)(uintptr_t)&sBhi[nb][0]);
      tdm_load_tile(gBlo + koff, (unsigned)(uintptr_t)&sBlo[nb][0]);
    }

    // ---- compute chunk n: 8 tiles/wave, 3 WMMAs each (hi*hi + hi*lo + lo*hi)
    const int arow = (wave * 16 + lr) * LROW + lg * 8;
    v16bf a_hi = load_frag(&sAhi[buf][arow]);
    v16bf a_lo = load_frag(&sAlo[buf][arow]);
#pragma unroll
    for (int t = 0; t < 8; ++t) {
      const int boff = (t * 16 + lr) * LROW + lg * 8;
      v16bf b_hi = load_frag(&sBhi[buf][boff]);
      v16bf b_lo = load_frag(&sBlo[buf][boff]);
      acc[t] = __builtin_amdgcn_wmma_f32_16x16x32_bf16(
          false, a_hi, false, b_hi, (short)0, acc[t], false, false);
      acc[t] = __builtin_amdgcn_wmma_f32_16x16x32_bf16(
          false, a_hi, false, b_lo, (short)0, acc[t], false, false);
      acc[t] = __builtin_amdgcn_wmma_f32_16x16x32_bf16(
          false, a_lo, false, b_hi, (short)0, acc[t], false, false);
    }
  }

  // ---- epilogue: d2 = sq_i + sq_j - 2g ; out = exp(-d2/1024) ----
  const float scale = -1.0f / (float)KDIM;
  float* outb = out + (size_t)b * MROWS * MROWS;
#pragma unroll
  for (int t = 0; t < 8; ++t) {
    const int j = j0 + t * 16 + lr;
    const float sj = sSqB[t * 16 + lr];
#pragma unroll
    for (int r = 0; r < 8; ++r) {
      const int irow = wave * 16 + lg * 8 + r;
      const float g  = acc[t][r];
      const float d2 = sSqA[irow] + sj - 2.0f * g;
      outb[(size_t)(i0 + irow) * MROWS + j] = __expf(scale * d2);
    }
  }
}

// ---------------------------------------------------------------------------
// Fallback (round-1 fused kernel) for when ws_size is too small.
// ---------------------------------------------------------------------------
__global__ __launch_bounds__(256, 2)
void spd_rbf_wmma_fused_kernel(const float* __restrict__ x,
                               float* __restrict__ out) {
  __shared__ __attribute__((aligned(16))) __bf16 sAhi[TILE * LROW];
  __shared__ __attribute__((aligned(16))) __bf16 sAlo[TILE * LROW];
  __shared__ __attribute__((aligned(16))) __bf16 sBhi[TILE * LROW];
  __shared__ __attribute__((aligned(16))) __bf16 sBlo[TILE * LROW];
  __shared__ float sSqPartA[256];
  __shared__ float sSqPartB[256];
  __shared__ float sSqA[TILE];
  __shared__ float sSqB[TILE];

  const int b  = blockIdx.y;
  const int tm = blockIdx.x >> 2;
  const int tn = blockIdx.x & 3;
  const int tid = threadIdx.x;

  const float* xb = x + (size_t)b * MROWS * KDIM;
  const int i0 = tm * TILE;
  const int j0 = tn * TILE;

  const int srow  = tid >> 1;
  const int shalf = tid & 1;
  const float* gA = xb + (size_t)(i0 + srow) * KDIM + shalf * 16;
  const float* gB = xb + (size_t)(j0 + srow) * KDIM + shalf * 16;

  float sqA = 0.f, sqB = 0.f;

  const int wave = tid >> 5;
  const int lane = tid & 31;
  const int lg   = lane >> 4;
  const int lr   = lane & 15;

  v8f acc[8];
#pragma unroll
  for (int t = 0; t < 8; ++t) acc[t] = {};

  for (int k0 = 0; k0 < KDIM; k0 += KC) {
    __syncthreads();
    {
      const float4* pa = (const float4*)(gA + k0);
      const float4* pb = (const float4*)(gB + k0);
      v8bf ah[2], al[2], bh[2], bl[2];
#pragma unroll
      for (int q = 0; q < 4; ++q) {
        float4 fa = pa[q];
        float4 fb = pb[q];
        float fav[4] = {fa.x, fa.y, fa.z, fa.w};
        float fbv[4] = {fb.x, fb.y, fb.z, fb.w};
#pragma unroll
        for (int e = 0; e < 4; ++e) {
          const int idx = q * 4 + e;
          float f = fav[e];
          __bf16 h = (__bf16)f;
          __bf16 l = (__bf16)(f - (float)h);
          ah[idx >> 3][idx & 7] = h;
          al[idx >> 3][idx & 7] = l;
          sqA += f * f;
          f = fbv[e];
          h = (__bf16)f;
          l = (__bf16)(f - (float)h);
          bh[idx >> 3][idx & 7] = h;
          bl[idx >> 3][idx & 7] = l;
          sqB += f * f;
        }
      }
      __bf16* dA = sAhi + srow * LROW + shalf * 16;
      __bf16* dB = sBhi + srow * LROW + shalf * 16;
      *(v8bf*)(dA)     = ah[0];
      *(v8bf*)(dA + 8) = ah[1];
      *(v8bf*)(dB)     = bh[0];
      *(v8bf*)(dB + 8) = bh[1];
      dA = sAlo + srow * LROW + shalf * 16;
      dB = sBlo + srow * LROW + shalf * 16;
      *(v8bf*)(dA)     = al[0];
      *(v8bf*)(dA + 8) = al[1];
      *(v8bf*)(dB)     = bl[0];
      *(v8bf*)(dB + 8) = bl[1];
    }
    __syncthreads();

    const __bf16* pAh = sAhi + (wave * 16 + lr) * LROW + lg * 8;
    const __bf16* pAl = sAlo + (wave * 16 + lr) * LROW + lg * 8;
    v16bf a_hi = load_frag(pAh);
    v16bf a_lo = load_frag(pAl);
#pragma unroll
    for (int t = 0; t < 8; ++t) {
      const int boff = (t * 16 + lr) * LROW + lg * 8;
      v16bf b_hi = load_frag(sBhi + boff);
      v16bf b_lo = load_frag(sBlo + boff);
      acc[t] = __builtin_amdgcn_wmma_f32_16x16x32_bf16(
          false, a_hi, false, b_hi, (short)0, acc[t], false, false);
      acc[t] = __builtin_amdgcn_wmma_f32_16x16x32_bf16(
          false, a_hi, false, b_lo, (short)0, acc[t], false, false);
      acc[t] = __builtin_amdgcn_wmma_f32_16x16x32_bf16(
          false, a_lo, false, b_hi, (short)0, acc[t], false, false);
    }
  }

  __syncthreads();
  sSqPartA[tid] = sqA;
  sSqPartB[tid] = sqB;
  __syncthreads();
  if (tid < TILE) {
    sSqA[tid] = sSqPartA[2 * tid] + sSqPartA[2 * tid + 1];
    sSqB[tid] = sSqPartB[2 * tid] + sSqPartB[2 * tid + 1];
  }
  __syncthreads();

  const float scale = -1.0f / (float)KDIM;
  float* outb = out + (size_t)b * MROWS * MROWS;
#pragma unroll
  for (int t = 0; t < 8; ++t) {
    const int j = j0 + t * 16 + lr;
    const float sj = sSqB[t * 16 + lr];
#pragma unroll
    for (int r = 0; r < 8; ++r) {
      const int irow = wave * 16 + lg * 8 + r;
      const float g  = acc[t][r];
      const float d2 = sSqA[irow] + sj - 2.0f * g;
      outb[(size_t)(i0 + irow) * MROWS + j] = __expf(scale * d2);
    }
  }
}

extern "C" void kernel_launch(void* const* d_in, const int* in_sizes, int n_in,
                              void* d_out, int out_size, void* d_ws, size_t ws_size,
                              hipStream_t stream) {
  (void)in_sizes; (void)n_in; (void)out_size;
  const float* x = (const float*)d_in[0];
  float* out = (float*)d_out;

  const size_t nelem   = (size_t)BATCH * MROWS * KDIM;
  const size_t ws_need = 2 * nelem * sizeof(__bf16)
                       + (size_t)BATCH * MROWS * sizeof(float);

  if (ws_size >= ws_need) {
    __bf16* xhi = (__bf16*)d_ws;
    __bf16* xlo = xhi + nelem;
    float*  sqp = (float*)(xlo + nelem);
    split_convert_kernel<<<(BATCH * MROWS) / 8, 256, 0, stream>>>(x, xhi, xlo, sqp);
    dim3 grid(16, BATCH, 1);
    spd_rbf_wmma_tdm_kernel<<<grid, 256, 0, stream>>>(xhi, xlo, sqp, out);
  } else {
    dim3 grid(16, BATCH, 1);
    spd_rbf_wmma_fused_kernel<<<grid, 256, 0, stream>>>(x, out);
  }
}
